// GroupedQueryAttention_14422500180121
// MI455X (gfx1250) — matrix operand order repaired and checked by reading, not compiled, not run on hardware
//
#include <hip/hip_runtime.h>
#include <cstdint>
#include <cstddef>

// ---------------------------------------------------------------- constants
constexpr int kS      = 2048;             // sequence length
constexpr int kD      = 2048;             // d_model
constexpr int kH      = 32;               // query heads
constexpr int kKVH    = 8;                // kv heads
constexpr int kDK     = 64;               // head dim
constexpr int kDKV    = kKVH * kDK;       // 512
constexpr int kREP    = kH / kKVH;        // 4

// ---------------------------------------------------------------- wmma types
typedef __attribute__((ext_vector_type(16))) __bf16 bf16x16;
typedef __attribute__((ext_vector_type(8)))  __bf16 bf16x8;
typedef __attribute__((ext_vector_type(8)))  float  f32x8;
typedef int v4i __attribute__((vector_size(16)));   // 16-byte int vector

#define AS1 __attribute__((address_space(1)))
#define AS3 __attribute__((address_space(3)))

#if defined(__has_builtin)
#if __has_builtin(__builtin_amdgcn_global_load_async_to_lds_b128) && \
    __has_builtin(__builtin_amdgcn_s_wait_asynccnt)
#define HAVE_ASYNC_LDS 1
#endif
#endif

__device__ __forceinline__ unsigned short f32_to_bf16_bits(float f) {
  unsigned int u = __builtin_bit_cast(unsigned int, f);
  u += 0x7FFFu + ((u >> 16) & 1u);                 // round-to-nearest-even
  return (unsigned short)(u >> 16);
}
__device__ __forceinline__ __bf16 f32_to_bf16(float f) {
  unsigned short b = f32_to_bf16_bits(f);
  return __builtin_bit_cast(__bf16, b);
}
__device__ __forceinline__ bf16x8 load_bf16x8(const unsigned short* p) {
  return __builtin_bit_cast(bf16x8, *reinterpret_cast<const uint4*>(p));
}
__device__ __forceinline__ bf16x16 pack16(bf16x8 lo, bf16x8 hi) {
  bf16x16 r;
#pragma unroll
  for (int i = 0; i < 8; ++i) { r[i] = lo[i]; r[8 + i] = hi[i]; }
  return r;
}
__device__ __forceinline__ f32x8 wmma_bf16(bf16x16 a, bf16x16 b, f32x8 c) {
  // D = A(16x32) x B(32x16) + C, f32 accumulate
  return __builtin_amdgcn_wmma_f32_16x16x32_bf16(false, a, false, b,
                                                 (short)0, c, false, false);
}
// convert two float4 (8 consecutive f32) into 8 bf16 fragment elements
__device__ __forceinline__ void cvt8(bf16x16& a, int base, float4 f0, float4 f1) {
  a[base + 0] = f32_to_bf16(f0.x); a[base + 1] = f32_to_bf16(f0.y);
  a[base + 2] = f32_to_bf16(f0.z); a[base + 3] = f32_to_bf16(f0.w);
  a[base + 4] = f32_to_bf16(f1.x); a[base + 5] = f32_to_bf16(f1.y);
  a[base + 6] = f32_to_bf16(f1.z); a[base + 7] = f32_to_bf16(f1.w);
}

// ---------------------------------------------------------------- converters
__global__ void cvt_f32_to_bf16(unsigned short* __restrict__ dst,
                                const float* __restrict__ src, int n) {
  int i = blockIdx.x * blockDim.x + threadIdx.x;
  if (i < n) dst[i] = f32_to_bf16_bits(src[i]);
}

// src: [K,N] f32 row-major  ->  dst: [N,K] bf16 row-major
__global__ void transpose_f32_to_bf16(unsigned short* __restrict__ dst,
                                      const float* __restrict__ src,
                                      int K, int N) {
  int i = blockIdx.x * blockDim.x + threadIdx.x;
  if (i < K * N) {
    int k = i / N, n = i - k * N;
    dst[(size_t)n * K + k] = f32_to_bf16_bits(src[i]);
  }
}

// ---------------------------------------------------------------- bf16 GEMM
// C[M,N] = A[M,K] * Bt[N,K]^T + bias[N], K must be 2048 here.
// 256-thread block = 8 waves; the block's 16-column B panel (contiguous
// 64 KB of Bt) is staged once into LDS — async global->LDS when available —
// and shared by all 8 waves (8x reduction in global B traffic).
enum { OUT_F32 = 0, OUT_BF16 = 1, OUT_BF16_T = 2 };

template <int MODE>
__global__ void gemm_bf16_kernel(const unsigned short* __restrict__ A,
                                 const unsigned short* __restrict__ Bt,
                                 const float* __restrict__ bias,
                                 void* __restrict__ C,
                                 int M, int N, int K) {
  __shared__ unsigned short bpanel[16 * 2048];   // 64 KB (K == 2048)

  const int t    = threadIdx.x;
  const int lane = t & 31;
  const int wave = t >> 5;
  const int m    = lane & 15;     // A row within tile / B column within tile
  const int kh   = lane >> 4;     // lane-half select
  const int col0 = blockIdx.x * 16;
  const int row0 = blockIdx.y * 128 + wave * 16;

  // ---- stage B panel: rows col0..col0+15 of Bt are contiguous 16*K bf16
  const unsigned short* gpanel = Bt + (size_t)col0 * K;
  const int chunks = (16 * K) / 8;               // 16-byte chunks
#if HAVE_ASYNC_LDS
  for (int c = t; c < chunks; c += 256) {
    __builtin_amdgcn_global_load_async_to_lds_b128(
        (AS1 v4i*)(gpanel + c * 8), (AS3 v4i*)(&bpanel[c * 8]), 0, 0);
  }
  __builtin_amdgcn_s_wait_asynccnt(0);
#else
  for (int c = t; c < chunks; c += 256) {
    *reinterpret_cast<uint4*>(&bpanel[c * 8]) =
        *reinterpret_cast<const uint4*>(gpanel + c * 8);
  }
#endif
  __syncthreads();

  const unsigned short* arow = A + (size_t)(row0 + m) * K;
  const unsigned short* brow = &bpanel[m * K];   // LDS

  f32x8 acc = {};
  for (int k0 = 0; k0 < K; k0 += 32) {
    __builtin_prefetch(arow + k0 + 512, 0, 1);   // global_prefetch_b8
    bf16x16 a = pack16(load_bf16x8(arow + k0 + kh * 8),
                       load_bf16x8(arow + k0 + 16 + kh * 8));
    bf16x16 b = pack16(load_bf16x8(brow + k0 + kh * 8),
                       load_bf16x8(brow + k0 + 16 + kh * 8));
    acc = wmma_bf16(a, b, acc);
  }

  const float bv = bias[col0 + m];
#pragma unroll
  for (int r = 0; r < 8; ++r) {
    const int rr = row0 + r + 8 * kh;     // C layout: VGPR r -> M = r + 8*kh
    const int cc = col0 + m;              //           N = lane & 15
    const float v = acc[r] + bv;
    if (MODE == OUT_F32) {
      ((float*)C)[(size_t)rr * N + cc] = v;
    } else if (MODE == OUT_BF16) {
      ((unsigned short*)C)[(size_t)rr * N + cc] = f32_to_bf16_bits(v);
    } else {  // transposed bf16 (for V^T)
      ((unsigned short*)C)[(size_t)cc * M + rr] = f32_to_bf16_bits(v);
    }
  }
}

// ---------------------------------------------------------------- scores
// attn[h,q,k] = mask ? (Q[h,q,:] . K[h/4,k,:]) / 8 : -1e9   (raw, pre-softmax)
__global__ void scores_kernel(const unsigned short* __restrict__ Qp,  // [S,2048]
                              const unsigned short* __restrict__ Kp,  // [S,512]
                              const unsigned char*  __restrict__ mask,// [S,S]
                              float* __restrict__ attn) {             // [H,S,S]
  const int lane = threadIdx.x & 31;
  const int wave = threadIdx.x >> 5;  // 0..7: strided over k-tiles
  const int m    = lane & 15;
  const int kh   = lane >> 4;
  const int q0   = blockIdx.x * 16;
  const int h    = blockIdx.y;
  const int kvh  = h / kREP;

  // Q tile fragments (d_k = 64 -> two 32-deep fragments), loaded once
  const unsigned short* arow = Qp + (size_t)(q0 + m) * kD + h * kDK;
  bf16x16 a0 = pack16(load_bf16x8(arow + kh * 8),      load_bf16x8(arow + 16 + kh * 8));
  bf16x16 a1 = pack16(load_bf16x8(arow + 32 + kh * 8), load_bf16x8(arow + 48 + kh * 8));

  for (int kt = wave; kt < kS / 16; kt += 8) {
    const int c0 = kt * 16;
    const unsigned short* brow = Kp + (size_t)(c0 + m) * kDKV + kvh * kDK;
    bf16x16 b0 = pack16(load_bf16x8(brow + kh * 8),      load_bf16x8(brow + 16 + kh * 8));
    bf16x16 b1 = pack16(load_bf16x8(brow + 32 + kh * 8), load_bf16x8(brow + 48 + kh * 8));
    f32x8 acc = {};
    acc = wmma_bf16(a0, b0, acc);
    acc = wmma_bf16(a1, b1, acc);
#pragma unroll
    for (int r = 0; r < 8; ++r) {
      const int qq = q0 + r + 8 * kh;
      const int kk = c0 + m;
      const float sc = acc[r] * 0.125f;   // 1/sqrt(64)
      const bool keep = mask[(size_t)qq * kS + kk] != 0;
      attn[((size_t)h * kS + qq) * kS + kk] = keep ? sc : -1e9f;
    }
  }
}

// ---------------------------------------------------------------- row softmax
__global__ void softmax_kernel(float* __restrict__ attn) {
  __shared__ float rowbuf[kS];
  __shared__ float red[256];
  float* p = attn + (size_t)blockIdx.x * kS;
  const int t = threadIdx.x;

  float lm = -3.4e38f;
  for (int i = t; i < kS; i += 256) { float v = p[i]; rowbuf[i] = v; lm = fmaxf(lm, v); }
  red[t] = lm;
  __syncthreads();
  for (int s2 = 128; s2 > 0; s2 >>= 1) {
    if (t < s2) red[t] = fmaxf(red[t], red[t + s2]);
    __syncthreads();
  }
  const float mx = red[0];
  __syncthreads();

  float ls = 0.f;
  for (int i = t; i < kS; i += 256) { float e = __expf(rowbuf[i] - mx); rowbuf[i] = e; ls += e; }
  red[t] = ls;
  __syncthreads();
  for (int s2 = 128; s2 > 0; s2 >>= 1) {
    if (t < s2) red[t] += red[t + s2];
    __syncthreads();
  }
  const float inv = 1.0f / red[0];
  for (int i = t; i < kS; i += 256) p[i] = rowbuf[i] * inv;
}

// ---------------------------------------------------------------- ctx = attn @ V
// Ctx[s, h*64+d] = sum_k attn[h,s,k] * V[h/4,k,d];  Vt is [512, S] bf16
__global__ void ctx_kernel(const float* __restrict__ attn,          // [H,S,S]
                           const unsigned short* __restrict__ Vt,   // [512,S]
                           unsigned short* __restrict__ Ctx) {      // [S,2048]
  const int lane = threadIdx.x & 31;
  const int wave = threadIdx.x >> 5;  // 0..3: which 16-wide d-slice
  const int m    = lane & 15;
  const int kh   = lane >> 4;
  const int q0   = blockIdx.x * 16;
  const int h    = blockIdx.y;
  const int kvh  = h / kREP;
  const int d0   = wave * 16;

  const float*          arow = attn + ((size_t)h * kS + q0 + m) * kS;
  const unsigned short* brow = Vt + (size_t)(kvh * kDK + d0 + m) * kS;

  f32x8 acc = {};
  for (int k0 = 0; k0 < kS; k0 += 32) {
    // A fragment: attn f32 -> bf16 in registers; 16-byte vector loads
    const float4* lo = reinterpret_cast<const float4*>(arow + k0 + kh * 8);
    const float4* hi = reinterpret_cast<const float4*>(arow + k0 + 16 + kh * 8);
    bf16x16 a;
    cvt8(a, 0, lo[0], lo[1]);
    cvt8(a, 8, hi[0], hi[1]);
    bf16x16 b = pack16(load_bf16x8(brow + k0 + kh * 8),
                       load_bf16x8(brow + k0 + 16 + kh * 8));
    acc = wmma_bf16(a, b, acc);
  }
#pragma unroll
  for (int r = 0; r < 8; ++r) {
    const int qq = q0 + r + 8 * kh;
    Ctx[(size_t)qq * kD + h * kDK + d0 + m] = f32_to_bf16_bits(acc[r]);
  }
}

// ---------------------------------------------------------------- launch
extern "C" void kernel_launch(void* const* d_in, const int* in_sizes, int n_in,
                              void* d_out, int out_size, void* d_ws, size_t ws_size,
                              hipStream_t stream) {
  (void)in_sizes; (void)n_in; (void)out_size; (void)ws_size;

  const float*         q    = (const float*)d_in[0];
  const float*         kin  = (const float*)d_in[1];
  const float*         vin  = (const float*)d_in[2];
  const unsigned char* mask = (const unsigned char*)d_in[3];
  const float*         wq   = (const float*)d_in[4];
  const float*         bq   = (const float*)d_in[5];
  const float*         wk   = (const float*)d_in[6];
  const float*         bk   = (const float*)d_in[7];
  const float*         wv   = (const float*)d_in[8];
  const float*         bv   = (const float*)d_in[9];
  const float*         wo   = (const float*)d_in[10];
  const float*         bo   = (const float*)d_in[11];

  char* ws = (char*)d_ws;
  const size_t MB = 1024 * 1024;
  unsigned short* Xq  = (unsigned short*)(ws + 0 * MB);   // [S,2048] bf16
  unsigned short* Xk  = (unsigned short*)(ws + 8 * MB);   // [S,2048]
  unsigned short* Xv  = (unsigned short*)(ws + 16 * MB);  // [S,2048]
  unsigned short* WqT = (unsigned short*)(ws + 24 * MB);  // [2048,2048]
  unsigned short* WkT = (unsigned short*)(ws + 32 * MB);  // [512,2048]
  unsigned short* WvT = (unsigned short*)(ws + 34 * MB);  // [512,2048]
  unsigned short* WoT = (unsigned short*)(ws + 36 * MB);  // [2048,2048]
  unsigned short* Qp  = (unsigned short*)(ws + 44 * MB);  // [S,2048]
  unsigned short* Kp  = (unsigned short*)(ws + 52 * MB);  // [S,512]
  unsigned short* Vt  = (unsigned short*)(ws + 54 * MB);  // [512,S]
  unsigned short* Ctx = (unsigned short*)(ws + 56 * MB);  // [S,2048]
  // total scratch: 64 MB

  float* out  = (float*)d_out;                    // [S,2048]
  float* attn = out + (size_t)kS * kD;            // [H,S,S]

  const int nSD = kS * kD;                        // 4,194,304
  cvt_f32_to_bf16<<<nSD / 256, 256, 0, stream>>>(Xq, q,   nSD);
  cvt_f32_to_bf16<<<nSD / 256, 256, 0, stream>>>(Xk, kin, nSD);
  cvt_f32_to_bf16<<<nSD / 256, 256, 0, stream>>>(Xv, vin, nSD);
  transpose_f32_to_bf16<<<nSD / 256, 256, 0, stream>>>(WqT, wq, kD, kD);
  transpose_f32_to_bf16<<<(kD * kDKV) / 256, 256, 0, stream>>>(WkT, wk, kD, kDKV);
  transpose_f32_to_bf16<<<(kD * kDKV) / 256, 256, 0, stream>>>(WvT, wv, kD, kDKV);
  transpose_f32_to_bf16<<<nSD / 256, 256, 0, stream>>>(WoT, wo, kD, kD);

  // projections: Q -> bf16 [S,2048], K -> bf16 [S,512], V -> bf16^T [512,S]
  gemm_bf16_kernel<OUT_BF16><<<dim3(kD / 16, kS / 128), 256, 0, stream>>>(
      Xq, WqT, bq, Qp, kS, kD, kD);
  gemm_bf16_kernel<OUT_BF16><<<dim3(kDKV / 16, kS / 128), 256, 0, stream>>>(
      Xk, WkT, bk, Kp, kS, kDKV, kD);
  gemm_bf16_kernel<OUT_BF16_T><<<dim3(kDKV / 16, kS / 128), 256, 0, stream>>>(
      Xv, WvT, bv, Vt, kS, kDKV, kD);

  // attention
  scores_kernel<<<dim3(kS / 16, kH), 256, 0, stream>>>(Qp, Kp, mask, attn);
  softmax_kernel<<<kH * kS, 256, 0, stream>>>(attn);
  ctx_kernel<<<dim3(kS / 16, kH), 128, 0, stream>>>(attn, Vt, Ctx);

  // out = ctx @ wo + bo (f32 output)
  gemm_bf16_kernel<OUT_F32><<<dim3(kD / 16, kS / 128), 256, 0, stream>>>(
      Ctx, WoT, bo, out, kS, kD, kD);
}
